// Decoder_34291018891502
// MI455X (gfx1250) — matrix operand order repaired and checked by reading, not compiled, not run on hardware
//
#include <hip/hip_runtime.h>
#include <hip/hip_bf16.h>

// ---------------------------------------------------------------------------
// Problem constants (from reference)
// ---------------------------------------------------------------------------
constexpr int B  = 64;
constexpr int S  = 256;
constexpr int T  = 48;
constexpr int H  = 512;
constexpr int NL = 32;
constexpr int NH = 2;
constexpr int DH = H / NH;          // 256
constexpr int K0 = NL + H;          // 544  (= 17 * 32, WMMA K-chunk friendly)
constexpr int H4 = 4 * H;           // 2048
constexpr float ATT_SCALE = 0.0625f; // 1/sqrt(DH)

typedef __attribute__((ext_vector_type(16))) __bf16 v16bf;
typedef __attribute__((ext_vector_type(8)))  __bf16 v8bf;
typedef __attribute__((ext_vector_type(8)))  float  v8f;

// ---------------------------------------------------------------------------
// f32 -> bf16 conversion
// ---------------------------------------------------------------------------
__global__ void cvt_bf16_kernel(const float* __restrict__ src,
                                __bf16* __restrict__ dst, int n) {
    int i = blockIdx.x * 256 + threadIdx.x;
    if (i < n) dst[i] = (__bf16)src[i];
}

// ---------------------------------------------------------------------------
// Init: c0=c1=cell, h0b=h1b=bf16(hidden)
// ---------------------------------------------------------------------------
__global__ void init_state_kernel(const float* __restrict__ hidden,
                                  const float* __restrict__ cell,
                                  float* __restrict__ c0, float* __restrict__ c1,
                                  __bf16* __restrict__ h0b, __bf16* __restrict__ h1b,
                                  int n) {
    int i = blockIdx.x * 256 + threadIdx.x;
    if (i < n) {
        float c = cell[i];
        c0[i] = c; c1[i] = c;
        __bf16 h = (__bf16)hidden[i];
        h0b[i] = h; h1b[i] = h;
    }
}

// ---------------------------------------------------------------------------
// Generic WMMA GEMM:  C[M,N] = A[M,K1] * B1[N,K1]^T  (+ A2[M,K2] * B2[N,K2]^T)
//                             + bias1[N] (+ bias2[N])
// Weights are [N,K] row-major (torch Linear layout) -> B-fragment loads are
// fully contiguous, no transpose needed.  Each wave computes a 16 x (16*NTC)
// strip: one A-fragment is reused by NTC consecutive WMMAs.  NTC is a
// template constant so every accumulator index is static (no v_movrel, no
// EXEC games around WMMA).  The wave id goes through readfirstlane so the
// group math and bounds check are scalar/uniform.  8 waves / 256-thr block.
// ---------------------------------------------------------------------------
__device__ __forceinline__ v16bf load_a_frag(const __bf16* base, int k0, int hi) {
    // A (16x32 bf16): lane&15 = row M; VGPR0-3 hold K = hi*16 + 0..7,
    // VGPR4-7 hold K = hi*16 + 8..15  -> two contiguous 16B loads.
    v8bf a0 = *(const v8bf*)(base + k0 + hi * 16);
    v8bf a1 = *(const v8bf*)(base + k0 + hi * 16 + 8);
    v16bf a;
#pragma unroll
    for (int i = 0; i < 8; ++i) { a[i] = a0[i]; a[8 + i] = a1[i]; }
    return a;
}

template <int NTC>
__global__ void wmma_gemm_kernel(const __bf16* __restrict__ A,  int K1,
                                 const __bf16* __restrict__ B1,
                                 const __bf16* __restrict__ A2, int K2,
                                 const __bf16* __restrict__ B2,
                                 const float* __restrict__ bias1,
                                 const float* __restrict__ bias2,
                                 float* __restrict__ outF,
                                 __bf16* __restrict__ outB,
                                 int ldc, int M, int N) {
    // wave id is uniform across the wave: make it explicitly scalar
    const int wave = __builtin_amdgcn_readfirstlane(threadIdx.x >> 5);
    const int lane = threadIdx.x & 31;
    const int tiles_n = N >> 4;
    const int tiles_m = M >> 4;
    const int ngroups = tiles_n / NTC;        // launch guarantees divisibility
    const int group   = blockIdx.x * 8 + wave;
    if (group >= tiles_m * ngroups) return;   // scalar branch; EXEC untouched
    const int tm  = group / ngroups;
    const int tn0 = (group % ngroups) * NTC;
    const int lo = lane & 15;
    const int hi = lane >> 4;

    v8f acc[NTC];
#pragma unroll
    for (int j = 0; j < NTC; ++j) acc[j] = (v8f){};

    // ---- first operand pair ----
    {
        const __bf16* aBase = A  + (size_t)(tm * 16 + lo) * K1;
        const __bf16* bBase = B1 + (size_t)((tn0 * 16) + lo) * K1;
        for (int k0 = 0; k0 < K1; k0 += 32) {
            v16bf a = load_a_frag(aBase, k0, hi);
#pragma unroll
            for (int j = 0; j < NTC; ++j) {
                // B (32x16): lane&15 = col; lanes<16 hold K=k0..k0+15,
                // lanes>=16 hold K=k0+16..k0+31 -> one contiguous 32B load.
                v16bf b = *(const v16bf*)(bBase + (size_t)j * 16 * K1 + k0 + hi * 16);
                acc[j] = __builtin_amdgcn_wmma_f32_16x16x32_bf16(
                    false, a, false, b, (short)0, acc[j], false, false);
            }
        }
    }
    // ---- optional second accumulation (e.g. h @ Whh^T) ----
    if (A2) {
        const __bf16* aBase = A2 + (size_t)(tm * 16 + lo) * K2;
        const __bf16* bBase = B2 + (size_t)((tn0 * 16) + lo) * K2;
        for (int k0 = 0; k0 < K2; k0 += 32) {
            v16bf a = load_a_frag(aBase, k0, hi);
#pragma unroll
            for (int j = 0; j < NTC; ++j) {
                v16bf b = *(const v16bf*)(bBase + (size_t)j * 16 * K2 + k0 + hi * 16);
                acc[j] = __builtin_amdgcn_wmma_f32_16x16x32_bf16(
                    false, a, false, b, (short)0, acc[j], false, false);
            }
        }
    }

    // ---- epilogue: lane&15 = col, VGPR r -> row = hi*8 + r ----
#pragma unroll
    for (int j = 0; j < NTC; ++j) {
        const int col = (tn0 + j) * 16 + lo;
        float bi = bias1 ? bias1[col] : 0.0f;
        if (bias2) bi += bias2[col];
#pragma unroll
        for (int r = 0; r < 8; ++r) {
            const int row = tm * 16 + hi * 8 + r;
            float v = acc[j][r] + bi;
            if (outF) outF[(size_t)row * ldc + col] = v;
            if (outB) outB[(size_t)row * ldc + col] = (__bf16)v;
        }
    }
}

// ---------------------------------------------------------------------------
// Per-step attention: one block per batch element (256 threads == S).
//   score[s]  = sum_h tanh(keys1[b,s,h] + q2[b,h]) * V[h] + bV   -> softmax -> aw
//   ctxsum[b,h] = sum_s aw[s] * enc[b,s,h]
//   logits[s,head] = (qh[b,head,:] . kenc[b,s,head,:]) * scale   (NO softmax: source bug)
//   mraw[b,h]   = sum_s logits[s, h/DH] * venc[b,s,h]
// ---------------------------------------------------------------------------
__device__ __forceinline__ float block_reduce(float v, float* red, int tid, bool do_max) {
    red[tid] = v;
    __syncthreads();
    for (int off = 128; off > 0; off >>= 1) {
        if (tid < off)
            red[tid] = do_max ? fmaxf(red[tid], red[tid + off]) : red[tid] + red[tid + off];
        __syncthreads();
    }
    float r = red[0];
    __syncthreads();
    return r;
}

__global__ void attn_step_kernel(const float* __restrict__ keys1,
                                 const float* __restrict__ q2,
                                 const float* __restrict__ qh,
                                 const float* __restrict__ Vw,
                                 const float* __restrict__ bV,
                                 const float* __restrict__ enc,
                                 const __bf16* __restrict__ kencb,
                                 const __bf16* __restrict__ vencb,
                                 float* __restrict__ ctxsum,
                                 __bf16* __restrict__ mrawb) {
    __shared__ float s_q2[H];
    __shared__ float s_qh[H];
    __shared__ float s_V[H];
    __shared__ float s_aw[S];
    __shared__ float s_l[S * NH];
    __shared__ float s_red[256];

    const int b = blockIdx.x;
    const int tid = threadIdx.x;

    for (int h = tid; h < H; h += 256) {
        s_q2[h] = q2[b * H + h];
        s_qh[h] = qh[b * H + h];
        s_V[h]  = Vw[h];
    }
    __syncthreads();

    const int s = tid;  // S == blockDim.x == 256
    const float*  kr   = keys1 + ((size_t)b * S + s) * H;
    const __bf16* krow = kencb + ((size_t)b * S + s) * H;

    float sc = 0.0f;
    float l0 = 0.0f, l1 = 0.0f;
    for (int h = 0; h < H; ++h) {
        if ((h & 63) == 0) __builtin_prefetch(kr + h + 128, 0, 0);  // global_prefetch_b8
        sc += tanhf(kr[h] + s_q2[h]) * s_V[h];
    }
    sc += bV[0];
    for (int d = 0; d < DH; ++d) {
        l0 += s_qh[d]      * (float)krow[d];
        l1 += s_qh[DH + d] * (float)krow[DH + d];
    }
    s_l[s * NH + 0] = l0 * ATT_SCALE;
    s_l[s * NH + 1] = l1 * ATT_SCALE;

    // softmax over s
    float mx  = block_reduce(sc, s_red, tid, true);
    float e   = expf(sc - mx);
    float sum = block_reduce(e, s_red, tid, false);
    s_aw[s] = e / sum;
    __syncthreads();

    // column reductions: ctx and mraw
    for (int h = tid; h < H; h += 256) {
        const int head = h >> 8;  // h / DH
        float cacc = 0.0f, macc = 0.0f;
        for (int si = 0; si < S; ++si) {
            const size_t idx = ((size_t)b * S + si) * H + h;
            cacc += s_aw[si] * enc[idx];
            macc += s_l[si * NH + head] * (float)vencb[idx];
        }
        ctxsum[b * H + h] = cacc;
        mrawb[b * H + h]  = (__bf16)macc;
    }
}

// ---------------------------------------------------------------------------
// x0 = concat(decoder_input[:,t,:], (ctxsum + mctx) * 0.5)  -> bf16 [B, K0]
// ---------------------------------------------------------------------------
__global__ void build_x0_kernel(const float* __restrict__ dec, int t,
                                const float* __restrict__ ctxsum,
                                const float* __restrict__ mctx,
                                __bf16* __restrict__ x0b) {
    int i = blockIdx.x * 256 + threadIdx.x;
    if (i >= B * K0) return;
    int b = i / K0, j = i % K0;
    float v;
    if (j < NL) v = dec[((size_t)b * T + t) * NL + j];
    else {
        int h = j - NL;
        v = (ctxsum[b * H + h] + mctx[b * H + h]) * 0.5f;
    }
    x0b[i] = (__bf16)v;
}

// ---------------------------------------------------------------------------
// LSTM cell activations + LayerNorm.  One block per batch row.
//   z[b, 0:H]=i, [H:2H]=f, [2H:3H]=g, [3H:4H]=o
//   c2 = sig(f)*c + sig(i)*tanh(g);  h2 = sig(o)*tanh(c2)
//   xn = LN(h2)*gamma + beta  (bf16, feeds next GEMM); h2 -> bf16 state
// ---------------------------------------------------------------------------
__device__ __forceinline__ float sigm(float x) { return 1.0f / (1.0f + expf(-x)); }

__global__ void lstm_act_kernel(const float* __restrict__ z,
                                float* __restrict__ cst,     // in/out cell state
                                const float* __restrict__ g,
                                const float* __restrict__ be,
                                __bf16* __restrict__ hb,     // bf16 hidden state out
                                __bf16* __restrict__ xnb) {  // bf16 LN output
    __shared__ float s_h[H];
    __shared__ float s_red[256];
    const int b = blockIdx.x;
    const int tid = threadIdx.x;

    for (int h = tid; h < H; h += 256) {
        const size_t zb = (size_t)b * H4;
        float zi = z[zb + h];
        float zf = z[zb + H + h];
        float zg = z[zb + 2 * H + h];
        float zo = z[zb + 3 * H + h];
        float c2 = sigm(zf) * cst[b * H + h] + sigm(zi) * tanhf(zg);
        float h2 = sigm(zo) * tanhf(c2);
        cst[b * H + h] = c2;
        hb[b * H + h]  = (__bf16)h2;
        s_h[h] = h2;
    }
    __syncthreads();

    float part = 0.0f;
    for (int h = tid; h < H; h += 256) part += s_h[h];
    float mean = block_reduce(part, s_red, tid, false) * (1.0f / H);

    part = 0.0f;
    for (int h = tid; h < H; h += 256) {
        float d = s_h[h] - mean;
        part += d * d;
    }
    float var  = block_reduce(part, s_red, tid, false) * (1.0f / H);
    float rstd = rsqrtf(var + 1e-5f);

    for (int h = tid; h < H; h += 256)
        xnb[b * H + h] = (__bf16)((s_h[h] - mean) * rstd * g[h] + be[h]);
}

// ---------------------------------------------------------------------------
// Host-side launch
// ---------------------------------------------------------------------------
static inline void cvt(hipStream_t st, const float* s, __bf16* d, int n) {
    cvt_bf16_kernel<<<(n + 255) / 256, 256, 0, st>>>(s, d, n);
}
static inline void gemm(hipStream_t st,
                        const __bf16* A, int K1, const __bf16* B1,
                        const __bf16* A2, int K2, const __bf16* B2,
                        const float* b1, const float* b2,
                        float* oF, __bf16* oB, int ldc, int M, int N) {
    int tiles_n = N / 16;
    if (tiles_n % 4 == 0) {
        int groups = (M / 16) * (tiles_n / 4);
        wmma_gemm_kernel<4><<<(groups + 7) / 8, 256, 0, st>>>(
            A, K1, B1, A2, K2, B2, b1, b2, oF, oB, ldc, M, N);
    } else if (tiles_n % 2 == 0) {
        int groups = (M / 16) * (tiles_n / 2);
        wmma_gemm_kernel<2><<<(groups + 7) / 8, 256, 0, st>>>(
            A, K1, B1, A2, K2, B2, b1, b2, oF, oB, ldc, M, N);
    } else {
        int groups = (M / 16) * tiles_n;
        wmma_gemm_kernel<1><<<(groups + 7) / 8, 256, 0, st>>>(
            A, K1, B1, A2, K2, B2, b1, b2, oF, oB, ldc, M, N);
    }
}

extern "C" void kernel_launch(void* const* d_in, const int* in_sizes, int n_in,
                              void* d_out, int out_size, void* d_ws, size_t ws_size,
                              hipStream_t stream) {
    // ---- inputs (setup_inputs order) ----
    const float* dec   = (const float*)d_in[0];   // [B,T,NL]
    const float* enc   = (const float*)d_in[1];   // [B,S,H]
    const float* hidden= (const float*)d_in[2];   // [B,H]
    const float* cell  = (const float*)d_in[3];   // [B,H]
    const float* W1    = (const float*)d_in[4];
    const float* b1    = (const float*)d_in[5];
    const float* W2    = (const float*)d_in[6];
    const float* b2    = (const float*)d_in[7];
    const float* Vw    = (const float*)d_in[8];   // [1,H]
    const float* bV    = (const float*)d_in[9];   // [1]
    const float* Wq    = (const float*)d_in[10];
    const float* bq    = (const float*)d_in[11];
    const float* Wv    = (const float*)d_in[12];
    const float* bv    = (const float*)d_in[13];
    const float* Wd    = (const float*)d_in[14];
    const float* bd    = (const float*)d_in[15];
    const float* Wih0  = (const float*)d_in[16];  // [4H, K0]
    const float* Whh0  = (const float*)d_in[17];  // [4H, H]
    const float* bih0  = (const float*)d_in[18];
    const float* bhh0  = (const float*)d_in[19];
    const float* Wih1  = (const float*)d_in[20];  // [4H, H]
    const float* Whh1  = (const float*)d_in[21];
    const float* bih1  = (const float*)d_in[22];
    const float* bhh1  = (const float*)d_in[23];
    const float* g0    = (const float*)d_in[24];
    const float* be0   = (const float*)d_in[25];
    const float* g1    = (const float*)d_in[26];
    const float* be1   = (const float*)d_in[27];
    const float* Wout  = (const float*)d_in[28];  // [NL, H]
    const float* bout  = (const float*)d_in[29];
    float* out = (float*)d_out;                   // [B, T*NL]

    // ---- workspace carve-up ----
    size_t off = 0;
    char* base = (char*)d_ws;
    auto alloc = [&](size_t bytes) -> void* {
        size_t o = (off + 255) & ~(size_t)255;
        off = o + bytes;
        return (void*)(base + o);
    };
    __bf16* W1b   = (__bf16*)alloc((size_t)H * H * 2);
    __bf16* Wvb   = (__bf16*)alloc((size_t)H * H * 2);
    __bf16* W2b   = (__bf16*)alloc((size_t)H * H * 2);
    __bf16* Wqb   = (__bf16*)alloc((size_t)H * H * 2);
    __bf16* Wdb   = (__bf16*)alloc((size_t)H * H * 2);
    __bf16* Wih0b = (__bf16*)alloc((size_t)H4 * K0 * 2);
    __bf16* Whh0b = (__bf16*)alloc((size_t)H4 * H * 2);
    __bf16* Wih1b = (__bf16*)alloc((size_t)H4 * H * 2);
    __bf16* Whh1b = (__bf16*)alloc((size_t)H4 * H * 2);
    __bf16* Woutb = (__bf16*)alloc((size_t)NL * H * 2);
    __bf16* encb  = (__bf16*)alloc((size_t)B * S * H * 2);
    __bf16* vencb = (__bf16*)alloc((size_t)B * S * H * 2);
    __bf16* kencb = (__bf16*)alloc((size_t)B * S * H * 2);
    float*  keys1 = (float*) alloc((size_t)B * S * H * 4);
    float*  q2    = (float*) alloc((size_t)B * H * 4);
    float*  qh    = (float*) alloc((size_t)B * H * 4);
    float*  ctxs  = (float*) alloc((size_t)B * H * 4);
    float*  mctx  = (float*) alloc((size_t)B * H * 4);
    __bf16* mrawb = (__bf16*)alloc((size_t)B * H * 2);
    __bf16* h0b   = (__bf16*)alloc((size_t)B * H * 2);
    __bf16* h1b   = (__bf16*)alloc((size_t)B * H * 2);
    float*  c0    = (float*) alloc((size_t)B * H * 4);
    float*  c1    = (float*) alloc((size_t)B * H * 4);
    __bf16* x0b   = (__bf16*)alloc((size_t)B * K0 * 2);
    __bf16* x1b   = (__bf16*)alloc((size_t)B * H * 2);
    __bf16* x2b   = (__bf16*)alloc((size_t)B * H * 2);
    float*  z0    = (float*) alloc((size_t)B * H4 * 4);
    float*  z1    = (float*) alloc((size_t)B * H4 * 4);
    (void)ws_size; (void)in_sizes; (void)n_in; (void)out_size;

    // ---- weight / input down-conversion to bf16 ----
    cvt(stream, W1,   W1b,   H * H);
    cvt(stream, Wv,   Wvb,   H * H);
    cvt(stream, W2,   W2b,   H * H);
    cvt(stream, Wq,   Wqb,   H * H);
    cvt(stream, Wd,   Wdb,   H * H);
    cvt(stream, Wih0, Wih0b, H4 * K0);
    cvt(stream, Whh0, Whh0b, H4 * H);
    cvt(stream, Wih1, Wih1b, H4 * H);
    cvt(stream, Whh1, Whh1b, H4 * H);
    cvt(stream, Wout, Woutb, NL * H);
    cvt(stream, enc,  encb,  B * S * H);

    // ---- loop-invariant precompute (WMMA) ----
    const int BS = B * S;
    // keys1 = enc @ W1^T + b1  (f32, reused 48x in the tanh-score loop)
    gemm(stream, encb, H, W1b, nullptr, 0, nullptr, b1, nullptr,
         keys1, nullptr, H, BS, H);
    // v_enc = enc @ Wv^T + bv  (bf16)
    gemm(stream, encb, H, Wvb, nullptr, 0, nullptr, bv, nullptr,
         nullptr, vencb, H, BS, H);
    // k_enc = v_enc @ Wv^T + bv  (source bug preserved; bf16)
    gemm(stream, vencb, H, Wvb, nullptr, 0, nullptr, bv, nullptr,
         nullptr, kencb, H, BS, H);

    init_state_kernel<<<(B * H + 255) / 256, 256, 0, stream>>>(
        hidden, cell, c0, c1, h0b, h1b, B * H);

    // ---- sequential scan over T steps ----
    for (int t = 0; t < T; ++t) {
        // projections of top-layer hidden
        gemm(stream, h1b, H, W2b, nullptr, 0, nullptr, b2, nullptr,
             q2, nullptr, H, B, H);
        gemm(stream, h1b, H, Wqb, nullptr, 0, nullptr, bq, nullptr,
             qh, nullptr, H, B, H);

        // attention (Bahdanau softmax ctx + raw-logit MHA branch)
        attn_step_kernel<<<B, 256, 0, stream>>>(
            keys1, q2, qh, Vw, bV, enc, kencb, vencb, ctxs, mrawb);

        // mctx = mraw @ Wd^T + bd
        gemm(stream, mrawb, H, Wdb, nullptr, 0, nullptr, bd, nullptr,
             mctx, nullptr, H, B, H);

        build_x0_kernel<<<(B * K0 + 255) / 256, 256, 0, stream>>>(
            dec, t, ctxs, mctx, x0b);

        // LSTM layer 0 gates: x0 @ Wih0^T + bih0 + h0 @ Whh0^T + bhh0
        gemm(stream, x0b, K0, Wih0b, h0b, H, Whh0b, bih0, bhh0,
             z0, nullptr, H4, B, H4);
        lstm_act_kernel<<<B, 256, 0, stream>>>(z0, c0, g0, be0, h0b, x1b);

        // LSTM layer 1 gates: x1 @ Wih1^T + bih1 + h1 @ Whh1^T + bhh1
        // (h1b still holds previous h1 here, as required)
        gemm(stream, x1b, H, Wih1b, h1b, H, Whh1b, bih1, bhh1,
             z1, nullptr, H4, B, H4);
        lstm_act_kernel<<<B, 256, 0, stream>>>(z1, c1, g1, be1, h1b, x2b);

        // out_t = LN(h1) @ Wout^T + bout  -> d_out[:, t*NL : (t+1)*NL]
        gemm(stream, x2b, H, Woutb, nullptr, 0, nullptr, bout, nullptr,
             out + t * NL, nullptr, T * NL, B, NL);
    }
}
